// MSDeformAttnTransformerEncoderLayer_82068235092330
// MI455X (gfx1250) — matrix operand order repaired and checked by reading, not compiled, not run on hardware
//
#include <hip/hip_runtime.h>
#include <hip/hip_bf16.h>
#include <math.h>

// ---------------------------------------------------------------------------
// Problem constants (match the reference)
// ---------------------------------------------------------------------------
#define BB      2
#define DD      256
#define NHH     8
#define HDD     32
#define NPP     4
#define DFFF    1024
#define LSRC_C  32400
#define LQ_C    300
#define NDET_C  200
#define LTOT_C  (LSRC_C + LQ_C)   // 32700
#define HSP     180
#define WSP     180

typedef __attribute__((ext_vector_type(16))) _Float16 v16h;
typedef __attribute__((ext_vector_type(8)))  float    v8f;
typedef __attribute__((ext_vector_type(4)))  float    f4;
typedef __attribute__((ext_vector_type(4)))  _Float16 h4;

// ---------------------------------------------------------------------------
// WMMA GEMM: Out[M,N] = act( X[M,K] @ W[N,K]^T + bias[N] )
// X row stride = ldx, Out row stride = ldo, W row stride = ldw (== K normally)
// ACT: 0=none 1=relu 2=gelu(exact) 3=sigmoid  (compile-time -> no branches)
// Block = 256 threads = 8 waves; tile = 64(M) x 64(N); K step = 32.
// LDS double-buffered: one barrier per K-step; global loads of tile k+1 are
// issued before the WMMAs of tile k so HBM latency hides behind matrix pipe.
// Staging loads are unconditional float4 on clamped row pointers with
// cndmask zeroing (no EXEC branching).
// ---------------------------------------------------------------------------
#define GTM 64
#define GTN 64
#define GTK 32
#define LDP (GTK + 8)   // padded LDS row stride in halves (80B, 16B aligned)

template <int ACT>
__global__ __launch_bounds__(256) void gemm_wmma_kernel(
    const float* __restrict__ X, int ldx,
    const float* __restrict__ W, int ldw,
    const float* __restrict__ bias,
    float* __restrict__ Out, int ldo,
    int M, int N, int K)
{
    __shared__ _Float16 Xs[2][GTM][LDP];
    __shared__ _Float16 Ws[2][GTN][LDP];

    const int tid  = threadIdx.x;
    const int wave = tid >> 5;
    const int lane = tid & 31;

    const int m0 = blockIdx.y * GTM;
    const int n0 = blockIdx.x * GTN;
    const int wm = wave >> 2;      // 0..1
    const int wn = wave & 3;       // 0..3

    // staging coordinates: thread t stages rows (t>>3) and (t>>3)+32,
    // cols (t&7)*4..+3 of each 64x32 tile.
    const int sr = tid >> 3;          // 0..31
    const int sc = (tid & 7) * 4;     // 0,4,...,28

    const int gm0 = m0 + sr,      gm1 = m0 + sr + 32;
    const int gn0 = n0 + sr,      gn1 = n0 + sr + 32;
    const bool vx0 = gm0 < M,     vx1 = gm1 < M;
    const bool vw0 = gn0 < N,     vw1 = gn1 < N;
    const float* const px0 = X + (size_t)(vx0 ? gm0 : (M - 1)) * ldx + sc;
    const float* const px1 = X + (size_t)(vx1 ? gm1 : (M - 1)) * ldx + sc;
    const float* const pw0 = W + (size_t)(vw0 ? gn0 : (N - 1)) * ldw + sc;
    const float* const pw1 = W + (size_t)(vw1 ? gn1 : (N - 1)) * ldw + sc;

    auto stage = [&](int buf, int k0) {
        const f4 x0 = *(const f4*)(px0 + k0);
        const f4 x1 = *(const f4*)(px1 + k0);
        const f4 w0 = *(const f4*)(pw0 + k0);
        const f4 w1 = *(const f4*)(pw1 + k0);
        h4 hx0, hx1, hw0, hw1;
        #pragma unroll
        for (int e = 0; e < 4; ++e) {
            hx0[e] = (_Float16)(vx0 ? x0[e] : 0.0f);
            hx1[e] = (_Float16)(vx1 ? x1[e] : 0.0f);
            hw0[e] = (_Float16)(vw0 ? w0[e] : 0.0f);
            hw1[e] = (_Float16)(vw1 ? w1[e] : 0.0f);
        }
        *(h4*)&Xs[buf][sr][sc]      = hx0;
        *(h4*)&Xs[buf][sr + 32][sc] = hx1;
        *(h4*)&Ws[buf][sr][sc]      = hw0;
        *(h4*)&Ws[buf][sr + 32][sc] = hw1;
        // prefetch two K-steps ahead (speculative; dropped on fault)
        if (k0 + 2 * GTK < K) {
            __builtin_prefetch((const void*)(px0 + k0 + 2 * GTK), 0, 0);
            __builtin_prefetch((const void*)(pw0 + k0 + 2 * GTK), 0, 0);
        }
    };

    stage(0, 0);
    __syncthreads();

    v8f acc0 = {};
    v8f acc1 = {};
    int cur = 0;

    for (int k0 = 0; k0 < K; k0 += GTK) {
        if (k0 + GTK < K) stage(cur ^ 1, k0 + GTK);

        // A fragments: lane<16 -> row=lane, K 0..7 & 16..23 ; lane>=16 -> K 8..15 & 24..31
        v16h a0, a1, b;
        const int arow = (lane & 15) + wm * 16;
        const int akb  = (lane >> 4) * 8;
        #pragma unroll
        for (int e = 0; e < 8; ++e) {
            a0[e]     = Xs[cur][arow][akb + e];
            a0[8 + e] = Xs[cur][arow][akb + 16 + e];
            a1[e]     = Xs[cur][arow + 32][akb + e];
            a1[8 + e] = Xs[cur][arow + 32][akb + 16 + e];
        }
        // B fragment: lane<16 -> col=lane, K 0..7 & 16..23 ; lane>=16 -> K 8..15 & 24..31
        const int bcol = (lane & 15) + wn * 16;
        const int bkb  = (lane >> 4) * 8;
        #pragma unroll
        for (int e = 0; e < 8; ++e) {
            b[e]     = Ws[cur][bcol][bkb + e];
            b[8 + e] = Ws[cur][bcol][bkb + 16 + e];
        }

        acc0 = __builtin_amdgcn_wmma_f32_16x16x32_f16(
                   false, a0, false, b, (short)0, acc0, false, false);
        acc1 = __builtin_amdgcn_wmma_f32_16x16x32_f16(
                   false, a1, false, b, (short)0, acc1, false, false);
        __syncthreads();
        cur ^= 1;
    }

    // ---- epilogue: lane&15 = N column, (lane>>4)*8 + v = M row ----
    const int ncol = n0 + wn * 16 + (lane & 15);
    if (ncol >= N) return;
    const float bv = bias[ncol];

    auto activate = [](float o) -> float {
        if (ACT == 1) o = fmaxf(o, 0.0f);
        if (ACT == 2) o = 0.5f * o * (1.0f + erff(o * 0.70710678118654752f));
        if (ACT == 3) o = __builtin_amdgcn_rcpf(1.0f + __expf(-o));
        return o;
    };

    const int rb = m0 + wm * 16 + (lane >> 4) * 8;
    if (m0 + GTM <= M) {
        #pragma unroll
        for (int v = 0; v < 8; ++v) {
            Out[(size_t)(rb + v) * ldo + ncol]      = activate(acc0[v] + bv);
            Out[(size_t)(rb + 32 + v) * ldo + ncol] = activate(acc1[v] + bv);
        }
    } else {
        #pragma unroll
        for (int v = 0; v < 8; ++v) {
            if (rb + v < M)
                Out[(size_t)(rb + v) * ldo + ncol] = activate(acc0[v] + bv);
            if (rb + 32 + v < M)
                Out[(size_t)(rb + 32 + v) * ldo + ncol] = activate(acc1[v] + bv);
        }
    }
}

// ---------------------------------------------------------------------------
// Fused residual add + LayerNorm over D=256. One block (256 thr) per row.
// ---------------------------------------------------------------------------
__global__ __launch_bounds__(256) void add_ln_kernel(
    const float* __restrict__ x, const float* __restrict__ res,
    const float* __restrict__ g, const float* __restrict__ b,
    float* __restrict__ out)
{
    const size_t row = blockIdx.x;
    const int t = threadIdx.x;
    float v = x[row * DD + t] + res[row * DD + t];

    __shared__ float red[8];
    float s = v;
    #pragma unroll
    for (int o = 16; o > 0; o >>= 1) s += __shfl_xor(s, o, 32);
    if ((t & 31) == 0) red[t >> 5] = s;
    __syncthreads();
    float tot = 0.0f;
    #pragma unroll
    for (int i = 0; i < 8; ++i) tot += red[i];
    const float mean = tot * (1.0f / DD);
    const float d = v - mean;
    __syncthreads();
    float s2 = d * d;
    #pragma unroll
    for (int o = 16; o > 0; o >>= 1) s2 += __shfl_xor(s2, o, 32);
    if ((t & 31) == 0) red[t >> 5] = s2;
    __syncthreads();
    float var = 0.0f;
    #pragma unroll
    for (int i = 0; i < 8; ++i) var += red[i];
    var *= (1.0f / DD);
    out[row * DD + t] = d * rsqrtf(var + 1e-5f) * g[t] + b[t];
}

// ---------------------------------------------------------------------------
// Elementwise helpers
// ---------------------------------------------------------------------------
__global__ void ew_add_kernel(const float* __restrict__ a,
                              const float* __restrict__ b,
                              float* __restrict__ c, int n)
{
    int i = blockIdx.x * blockDim.x + threadIdx.x;
    if (i < n) c[i] = a[i] + b[i];
}
__global__ void ew_mul_kernel(const float* __restrict__ a,
                              const float* __restrict__ b,
                              float* __restrict__ c, int n)
{
    int i = blockIdx.x * blockDim.x + threadIdx.x;
    if (i < n) c[i] = a[i] * b[i];
}

// ---------------------------------------------------------------------------
// MSDeformAttn sampling core. One wave per (b, query, head); lane = channel.
// value: [B, Lin, 256] (head h at cols h*32..h*32+31)
// off:   [B*Lq, 64], awl (logits): [B*Lq, 32], ref: [B, Ltot, 2]
// out:   [B*Lq, 256]
// ---------------------------------------------------------------------------
__global__ __launch_bounds__(256) void msda_sample_kernel(
    const float* __restrict__ value,
    const float* __restrict__ off,
    const float* __restrict__ awl,
    const float* __restrict__ ref,
    int ref_off, int Lq, int q_lo, int q_hi,
    float* __restrict__ out)
{
    const int wave = threadIdx.x >> 5;
    const int lane = threadIdx.x & 31;
    const int qc = q_hi - q_lo;
    const long u = (long)blockIdx.x * 8 + wave;
    const long total = (long)BB * qc * NHH;
    if (u >= total) return;

    const int b   = (int)(u / ((long)qc * NHH));
    const int rem = (int)(u % ((long)qc * NHH));
    const int qi  = q_lo + rem / NHH;
    const int h   = rem % NHH;
    const long row = (long)b * Lq + qi;

    const float rx = ref[((long)b * LTOT_C + ref_off + qi) * 2 + 0];
    const float ry = ref[((long)b * LTOT_C + ref_off + qi) * 2 + 1];

    // softmax over the NL*NP = 4 attention logits of this head
    float lg[NPP];
    float mx = -1e30f;
    #pragma unroll
    for (int p = 0; p < NPP; ++p) {
        lg[p] = awl[row * 32 + h * NPP + p];
        mx = fmaxf(mx, lg[p]);
    }
    float se = 0.0f;
    #pragma unroll
    for (int p = 0; p < NPP; ++p) { lg[p] = __expf(lg[p] - mx); se += lg[p]; }
    const float inv = 1.0f / se;

    const float* vbase = value + ((long)b * LSRC_C) * DD + h * HDD + lane;
    float acc = 0.0f;
    #pragma unroll
    for (int p = 0; p < NPP; ++p) {
        const float ox = off[row * 64 + (h * NPP + p) * 2 + 0];
        const float oy = off[row * 64 + (h * NPP + p) * 2 + 1];
        const float x = (rx + ox / (float)WSP) * (float)WSP - 0.5f;
        const float y = (ry + oy / (float)HSP) * (float)HSP - 0.5f;
        const float x0 = floorf(x), y0 = floorf(y);
        const float lx = x - x0, ly = y - y0;
        const int x0i = (int)x0, y0i = (int)y0;
        const float wgt[4] = { (1.0f - lx) * (1.0f - ly), lx * (1.0f - ly),
                               (1.0f - lx) * ly,          lx * ly };
        const int dxs[4] = {0, 1, 0, 1};
        const int dys[4] = {0, 0, 1, 1};
        float s = 0.0f;
        #pragma unroll
        for (int c = 0; c < 4; ++c) {
            const int xi = x0i + dxs[c];
            const int yi = y0i + dys[c];
            const bool valid = (xi >= 0) && (xi < WSP) && (yi >= 0) && (yi < HSP);
            const int xc = min(max(xi, 0), WSP - 1);
            const int yc = min(max(yi, 0), HSP - 1);
            const long idx = (long)yc * WSP + xc;
            const float gv = vbase[idx * DD];
            s += gv * (valid ? wgt[c] : 0.0f);
        }
        acc += s * lg[p] * inv;
    }
    out[row * DD + h * HDD + lane] = acc;
}

// ---------------------------------------------------------------------------
// Small MHA over Lq=300 queries. One wave per (b, h, q); lane = channel.
// Grid must cover exactly B*NH*Lq / 8 blocks (divisible: 4800/8 = 600).
// ---------------------------------------------------------------------------
__global__ __launch_bounds__(256) void mha_attn_kernel(
    const float* __restrict__ Q, const float* __restrict__ Kp,
    const float* __restrict__ Vp, float* __restrict__ O)
{
    __shared__ float sc[8][LQ_C];
    __shared__ float qrow[8][HDD];

    const int wave = threadIdx.x >> 5;
    const int lane = threadIdx.x & 31;
    const long u = (long)blockIdx.x * 8 + wave;
    const int b = (int)(u / (NHH * LQ_C));
    const int rem = (int)(u % (NHH * LQ_C));
    const int h = rem / LQ_C;
    const int q = rem % LQ_C;

    qrow[wave][lane] = Q[((size_t)(b * LQ_C + q) * DD) + h * HDD + lane];
    __syncthreads();

    const float scale = 0.17677669529663687f;   // 1/sqrt(32)
    for (int k = lane; k < LQ_C; k += 32) {
        const float* kptr = Kp + ((size_t)(b * LQ_C + k) * DD) + h * HDD;
        float d = 0.0f;
        #pragma unroll
        for (int e = 0; e < HDD; ++e) d += qrow[wave][e] * kptr[e];
        sc[wave][k] = d * scale;
    }
    // per-wave softmax (each lane only touches k = lane + 32*j entries)
    float mx = -1e30f;
    for (int k = lane; k < LQ_C; k += 32) mx = fmaxf(mx, sc[wave][k]);
    #pragma unroll
    for (int o = 16; o > 0; o >>= 1) mx = fmaxf(mx, __shfl_xor(mx, o, 32));
    float sum = 0.0f;
    for (int k = lane; k < LQ_C; k += 32) {
        float e = __expf(sc[wave][k] - mx);
        sc[wave][k] = e;
        sum += e;
    }
    #pragma unroll
    for (int o = 16; o > 0; o >>= 1) sum += __shfl_xor(sum, o, 32);
    __syncthreads();

    const float inv = 1.0f / sum;
    float acc = 0.0f;
    for (int k = 0; k < LQ_C; ++k)
        acc += sc[wave][k] * Vp[((size_t)(b * LQ_C + k) * DD) + h * HDD + lane];
    O[((size_t)(b * LQ_C + q) * DD) + h * HDD + lane] = acc * inv;
}

// ---------------------------------------------------------------------------
// Host-side orchestration
// ---------------------------------------------------------------------------
static inline void launch_gemm(hipStream_t s,
                               const float* X, int ldx,
                               const float* W, int ldw,
                               const float* bias,
                               float* Out, int ldo,
                               int M, int N, int K, int act)
{
    dim3 grid((N + GTN - 1) / GTN, (M + GTM - 1) / GTM);
    switch (act) {
    case 1:  gemm_wmma_kernel<1><<<grid, 256, 0, s>>>(X, ldx, W, ldw, bias, Out, ldo, M, N, K); break;
    case 2:  gemm_wmma_kernel<2><<<grid, 256, 0, s>>>(X, ldx, W, ldw, bias, Out, ldo, M, N, K); break;
    case 3:  gemm_wmma_kernel<3><<<grid, 256, 0, s>>>(X, ldx, W, ldw, bias, Out, ldo, M, N, K); break;
    default: gemm_wmma_kernel<0><<<grid, 256, 0, s>>>(X, ldx, W, ldw, bias, Out, ldo, M, N, K); break;
    }
}

extern "C" void kernel_launch(void* const* d_in, const int* in_sizes, int n_in,
                              void* d_out, int out_size, void* d_ws, size_t ws_size,
                              hipStream_t stream)
{
    (void)in_sizes; (void)n_in; (void)out_size; (void)ws_size;

    // ---- inputs (setup_inputs dict order) ----
    const float* query     = (const float*)d_in[0];
    const float* query_pos = (const float*)d_in[1];
    const float* src       = (const float*)d_in[2];
    const float* pos       = (const float*)d_in[3];
    const float* refpts    = (const float*)d_in[4];   // [B, 32700, 1, 2]
    // d_in[5] spatial_shapes, d_in[6] level_start_index: constants baked in
    const float* sa_Wv   = (const float*)d_in[7];  const float* sa_bv   = (const float*)d_in[8];
    const float* sa_Woff = (const float*)d_in[9];  const float* sa_boff = (const float*)d_in[10];
    const float* sa_Waw  = (const float*)d_in[11]; const float* sa_baw  = (const float*)d_in[12];
    const float* sa_Wout = (const float*)d_in[13]; const float* sa_bout = (const float*)d_in[14];
    const float* ca_Wv   = (const float*)d_in[15]; const float* ca_bv   = (const float*)d_in[16];
    const float* ca_Woff = (const float*)d_in[17]; const float* ca_boff = (const float*)d_in[18];
    const float* ca_Waw  = (const float*)d_in[19]; const float* ca_baw  = (const float*)d_in[20];
    const float* ca_Wout = (const float*)d_in[21]; const float* ca_bout = (const float*)d_in[22];
    const float* norm1_g = (const float*)d_in[23]; const float* norm1_b = (const float*)d_in[24];
    const float* norm2_g = (const float*)d_in[25]; const float* norm2_b = (const float*)d_in[26];
    const float* normq_g = (const float*)d_in[27]; const float* normq_b = (const float*)d_in[28];
    const float* qan_g   = (const float*)d_in[29]; const float* qan_b   = (const float*)d_in[30];
    const float* norm3_g = (const float*)d_in[31]; const float* norm3_b = (const float*)d_in[32];
    const float* ffn1_W1 = (const float*)d_in[33]; const float* ffn1_b1 = (const float*)d_in[34];
    const float* ffn1_W2 = (const float*)d_in[35]; const float* ffn1_b2 = (const float*)d_in[36];
    const float* ffn3_W1 = (const float*)d_in[37]; const float* ffn3_b1 = (const float*)d_in[38];
    const float* ffn3_W2 = (const float*)d_in[39]; const float* ffn3_b2 = (const float*)d_in[40];
    const float* tcs_Wp  = (const float*)d_in[41]; const float* tcs_bp  = (const float*)d_in[42];
    const float* tcs_Wg1 = (const float*)d_in[43]; const float* tcs_bg1 = (const float*)d_in[44];
    const float* tcs_Wg2 = (const float*)d_in[45]; const float* tcs_bg2 = (const float*)d_in[46];
    const float* qa_Win  = (const float*)d_in[47]; const float* qa_bin  = (const float*)d_in[48];
    const float* qa_Wout = (const float*)d_in[49]; const float* qa_bout = (const float*)d_in[50];

    // ---- outputs: (query, src, gate_det, gate_seg) flat ----
    const size_t MS = (size_t)BB * LSRC_C;   // 64800
    const size_t MQ = (size_t)BB * LQ_C;     // 600
    float* out_query = (float*)d_out;
    float* out_src   = out_query + MQ * DD;
    float* out_gd    = out_src   + MS * DD;
    float* out_gs    = out_gd    + MS * DD;

    // ---- workspace carve ----
    float* W0 = (float*)d_ws;
    float* S0 = W0;               // MS*256
    float* S1 = S0 + MS * DD;     // MS*256
    float* S2 = S1 + MS * DD;     // MS*256
    float* S3 = S2 + MS * DD;     // MS*256
    float* S4 = S3 + MS * DD;     // MS*1024 (FFN hidden / gelu gates, ld=512)
    float* S5 = S4 + MS * DFFF;   // MS*64  (sa offsets)
    float* S6 = S5 + MS * 64;     // MS*32  (sa aw logits)
    float* Qp = S6 + MS * 32;     // query-scale pool
    float* qq      = Qp;                 Qp += MQ * DD;
    float* off_ca  = Qp;                 Qp += MQ * 64;
    float* awl_ca  = Qp;                 Qp += MQ * 32;
    float* samp_ca = Qp;                 Qp += MQ * DD;
    float* query2  = Qp;                 Qp += MQ * DD;
    float* bufQ1   = Qp;                 Qp += MQ * DD;
    float* qk      = Qp;                 Qp += MQ * DD;
    float* qproj   = Qp;                 Qp += MQ * DD;
    float* kproj   = Qp;                 Qp += MQ * DD;
    float* vproj   = Qp;                 Qp += MQ * DD;
    float* mhout   = Qp;                 Qp += MQ * DD;
    float* mh      = Qp;                 Qp += MQ * DD;
    float* bufQ2   = Qp;                 Qp += MQ * DD;
    float* h3a     = Qp;                 Qp += MQ * DFFF;
    float* h3      = Qp;                 Qp += MQ * DD;

    const int nsrc = (int)(MS * DD);
    const int nq   = (int)(MQ * DD);

    // ================= self-attention (MSDA over src) =================
    ew_add_kernel<<<(nsrc + 255) / 256, 256, 0, stream>>>(src, pos, S0, nsrc); // qsrc
    launch_gemm(stream, src, DD, sa_Wv, DD, sa_bv, S1, DD, (int)MS, DD, DD, 0);      // value
    launch_gemm(stream, S0, DD, sa_Woff, DD, sa_boff, S5, 64, (int)MS, 64, DD, 0);   // offsets
    launch_gemm(stream, S0, DD, sa_Waw, DD, sa_baw, S6, 32, (int)MS, 32, DD, 0);     // aw logits
    {
        long units = (long)BB * LSRC_C * NHH;
        msda_sample_kernel<<<(unsigned)((units + 7) / 8), 256, 0, stream>>>(
            S1, S5, S6, refpts, /*ref_off=*/0, /*Lq=*/LSRC_C, 0, LSRC_C, S2);
    }
    launch_gemm(stream, S2, DD, sa_Wout, DD, sa_bout, S3, DD, (int)MS, DD, DD, 0);   // src2
    add_ln_kernel<<<(unsigned)MS, 256, 0, stream>>>(S3, src, norm1_g, norm1_b, S0);  // srcA

    // ================= FFN1 =================
    launch_gemm(stream, S0, DD, ffn1_W1, DD, ffn1_b1, S4, DFFF, (int)MS, DFFF, DD, 1);
    launch_gemm(stream, S4, DFFF, ffn1_W2, DFFF, ffn1_b2, S1, DD, (int)MS, DD, DFFF, 0);
    add_ln_kernel<<<(unsigned)MS, 256, 0, stream>>>(S1, S0, norm2_g, norm2_b, out_src); // final src

    // ================= task-conditioned gates =================
    launch_gemm(stream, out_src, DD, tcs_Wp, DD, tcs_bp, S4, 2 * DD, (int)MS, 2 * DD, DD, 2); // gelu
    launch_gemm(stream, S4, 2 * DD, tcs_Wg1, DD, tcs_bg1, out_gd, DD, (int)MS, DD, DD, 3);     // gate_det
    launch_gemm(stream, S4 + DD, 2 * DD, tcs_Wg2, DD, tcs_bg2, out_gs, DD, (int)MS, DD, DD, 3);// gate_seg

    // gated value projections for cross-attention
    ew_mul_kernel<<<(nsrc + 255) / 256, 256, 0, stream>>>(out_src, out_gd, S2, nsrc);
    launch_gemm(stream, S2, DD, ca_Wv, DD, ca_bv, S1, DD, (int)MS, DD, DD, 0);  // val_det
    ew_mul_kernel<<<(nsrc + 255) / 256, 256, 0, stream>>>(out_src, out_gs, S2, nsrc);
    launch_gemm(stream, S2, DD, ca_Wv, DD, ca_bv, S0, DD, (int)MS, DD, DD, 0);  // val_seg

    // ================= cross-attention (queries) =================
    ew_add_kernel<<<(nq + 255) / 256, 256, 0, stream>>>(query, query_pos, qq, nq);
    launch_gemm(stream, qq, DD, ca_Woff, DD, ca_boff, off_ca, 64, (int)MQ, 64, DD, 0);
    launch_gemm(stream, qq, DD, ca_Waw, DD, ca_baw, awl_ca, 32, (int)MQ, 32, DD, 0);
    {
        long ud = (long)BB * NDET_C * NHH;
        msda_sample_kernel<<<(unsigned)((ud + 7) / 8), 256, 0, stream>>>(
            S1, off_ca, awl_ca, refpts, /*ref_off=*/LSRC_C, LQ_C, 0, NDET_C, samp_ca);
        long us = (long)BB * (LQ_C - NDET_C) * NHH;
        msda_sample_kernel<<<(unsigned)((us + 7) / 8), 256, 0, stream>>>(
            S0, off_ca, awl_ca, refpts, /*ref_off=*/LSRC_C, LQ_C, NDET_C, LQ_C, samp_ca);
    }
    launch_gemm(stream, samp_ca, DD, ca_Wout, DD, ca_bout, query2, DD, (int)MQ, DD, DD, 0);
    add_ln_kernel<<<(unsigned)MQ, 256, 0, stream>>>(query2, query, normq_g, normq_b, bufQ1);

    // ================= query self-attention (MHA) =================
    ew_add_kernel<<<(nq + 255) / 256, 256, 0, stream>>>(bufQ1, query_pos, qk, nq);
    launch_gemm(stream, qk,    DD, qa_Win,               DD, qa_bin,          qproj, DD, (int)MQ, DD, DD, 0);
    launch_gemm(stream, qk,    DD, qa_Win + 1 * DD * DD, DD, qa_bin + 1 * DD, kproj, DD, (int)MQ, DD, DD, 0);
    launch_gemm(stream, bufQ1, DD, qa_Win + 2 * DD * DD, DD, qa_bin + 2 * DD, vproj, DD, (int)MQ, DD, DD, 0);
    mha_attn_kernel<<<(BB * NHH * LQ_C) / 8, 256, 0, stream>>>(qproj, kproj, vproj, mhout);
    launch_gemm(stream, mhout, DD, qa_Wout, DD, qa_bout, mh, DD, (int)MQ, DD, DD, 0);
    add_ln_kernel<<<(unsigned)MQ, 256, 0, stream>>>(mh, bufQ1, qan_g, qan_b, bufQ2);

    // ================= FFN3 =================
    launch_gemm(stream, bufQ2, DD, ffn3_W1, DD, ffn3_b1, h3a, DFFF, (int)MQ, DFFF, DD, 1);
    launch_gemm(stream, h3a, DFFF, ffn3_W2, DFFF, ffn3_b2, h3, DD, (int)MQ, DD, DFFF, 0);
    add_ln_kernel<<<(unsigned)MQ, 256, 0, stream>>>(h3, bufQ2, norm3_g, norm3_b, out_query);
}
